// EquivariantProductBasisBlock_3959959847294
// MI455X (gfx1250) — matrix operand order repaired and checked by reading, not compiled, not run on hardware
//
#include <hip/hip_runtime.h>
#include <stdint.h>

// ---------------------------------------------------------------------------
// MACE EquivariantProductBasisBlock for MI455X (gfx1250, wave32, WMMA).
//
// - Fold element weights into per-element bf16 tables (L2-resident, 27.5 MB).
// - Group nodes by species so the nu=3 contraction becomes dense per-(e,c)
//   16x16x32 bf16 WMMA GEMMs (A pre-packed in the ISA VGPR layout).
// - x rows staged global->LDS with GLOBAL_LOAD_ASYNC_TO_LDS_B32 (ASYNCcnt).
// - Lower-order Horner steps fused in LDS; final e3nn Linear = bf16 WMMA GEMM.
// ---------------------------------------------------------------------------

#define N_NODES 1024
#define C_CH    128
#define K_SPH   9
#define E_ELEM  10
#define MIJ     324          // (M0 + M1) * 81 = 4*81
#define MIJ_T   21           // ceil(324/16) row tiles (padded to 336)
#define MI_TOT  36           // (M0 + M1) * 9
#define LIST_CAP  1280

typedef __attribute__((ext_vector_type(16))) __bf16 v16bf;
typedef __attribute__((ext_vector_type(8)))  float  v8f;

// ---- workspace layout (bytes) ---------------------------------------------
#define APACK_SZ  (E_ELEM*C_CH*MIJ_T*32*16*2)   // 27,525,120 B, wmma-A packed bf16
#define APACK_OFF 0
#define C2W_OFF   (APACK_OFF + APACK_SZ)
#define C2W_SZ    (E_ELEM*C_CH*MIJ*4)           // 1,658,880 B
#define C1W_OFF   (C2W_OFF + C2W_SZ)
#define C1W_SZ    (E_ELEM*C_CH*MI_TOT*4)        // 184,320 B
#define Y_OFF     (C1W_OFF + C1W_SZ)
#define Y_SZ      (N_NODES*C_CH*4*4)            // y[n][c][m] f32, 2 MB
#define META_OFF  (Y_OFF + Y_SZ)
// meta ints: [0]=numTiles, [16..95]=tileElem, [96..175]=tileBase, [176..]=list

__device__ __forceinline__ unsigned short f2bf(float f) {
    union { float f; unsigned u; } v; v.f = f;
    unsigned r = v.u + 0x7FFFu + ((v.u >> 16) & 1u);   // round-to-nearest-even
    return (unsigned short)(r >> 16);
}

// ---------------------------------------------------------------------------
// K0: species extraction + grouping (single 1024-thread block, one per node).
// ---------------------------------------------------------------------------
__global__ __launch_bounds__(1024)
void k0_group(const float* __restrict__ na, int* __restrict__ meta) {
    __shared__ int cnt[E_ELEM], offs[E_ELEM], cursor[E_ELEM];
    int tid = threadIdx.x;
    int* tileElem = meta + 16;
    int* tileBase = meta + 96;
    int* list     = meta + 176;
    if (tid < E_ELEM) { cnt[tid] = 0; cursor[tid] = 0; }
    for (int i = tid; i < LIST_CAP; i += 1024) list[i] = -1;
    __syncthreads();
    int s = 0;
    const float* row = na + (size_t)tid * E_ELEM;
    for (int e = 0; e < E_ELEM; ++e) if (row[e] > 0.5f) s = e;   // one-hot
    atomicAdd(&cnt[s], 1);
    __syncthreads();
    if (tid == 0) {
        int running = 0, tiles = 0;
        for (int e = 0; e < E_ELEM; ++e) {
            offs[e] = running;
            int nt = (cnt[e] + 15) >> 4;
            for (int t = 0; t < nt; ++t) {
                tileElem[tiles] = e;
                tileBase[tiles] = running + t * 16;
                ++tiles;
            }
            running += nt * 16;
        }
        meta[0] = tiles;   // <= 74
    }
    __syncthreads();
    int pos = offs[s] + atomicAdd(&cursor[s], 1);
    list[pos] = tid;
}

// ---------------------------------------------------------------------------
// K1a: C3w[e,c,mij,k] = sum_p U3[i,j,k,m,p]*W3[e,p,c], pre-packed in the
// 16-bit WMMA A-matrix VGPR layout (16 rows x 32 K, K 9..31 zero-padded).
// ---------------------------------------------------------------------------
__global__ __launch_bounds__(256)
void k1_packA(const float* __restrict__ U3_0, const float* __restrict__ W3_0,
              const float* __restrict__ U3_1, const float* __restrict__ W3_1,
              unsigned short* __restrict__ apack) {
    int t = blockIdx.x * blockDim.x + threadIdx.x;
    if (t >= E_ELEM * C_CH * MIJ_T * 32) return;
    int lane = t & 31;
    int mt   = (t >> 5) % MIJ_T;
    int c    = ((t >> 5) / MIJ_T) % C_CH;
    int e    = (t >> 5) / (MIJ_T * C_CH);
    int row  = mt * 16 + (lane & 15);
    unsigned short* out = apack + (size_t)t * 16;
    #pragma unroll
    for (int s = 0; s < 16; ++s) {
        // A-layout: lanes 0-15: slots 0-7 -> K 0-7, slots 8-15 -> K 16-23
        //           lanes 16-31: slots 0-7 -> K 8-15, slots 8-15 -> K 24-31
        int k = (lane < 16) ? ((s < 8) ? s : 16 + (s - 8))
                            : ((s < 8) ? 8 + s : 24 + (s - 8));
        float v = 0.f;
        if (row < MIJ && k < K_SPH) {
            if (row < 81) {                       // L = 0, M = 1
                int i = row / 9, j = row % 9;
                int ub = (i * 81 + j * 9 + k) * 3;
                for (int p = 0; p < 3; ++p)
                    v += U3_0[ub + p] * W3_0[(e * 3 + p) * C_CH + c];
            } else {                              // L = 1, M = 3
                int q = row - 81;
                int m = q / 81, r2 = q % 81;
                int i = r2 / 9, j = r2 % 9;
                int ub = ((i * 81 + j * 9 + k) * 3 + m) * 3;
                for (int p = 0; p < 3; ++p)
                    v += U3_1[ub + p] * W3_1[(e * 3 + p) * C_CH + c];
            }
        }
        out[s] = f2bf(v);
    }
}

// K1b: C2w[e,c,mij] = sum_p U2[i,j,m,p] * W2[e,p,c]
__global__ __launch_bounds__(256)
void k1_c2(const float* __restrict__ U2_0, const float* __restrict__ W2_0,
           const float* __restrict__ U2_1, const float* __restrict__ W2_1,
           float* __restrict__ c2w) {
    int t = blockIdx.x * blockDim.x + threadIdx.x;
    if (t >= E_ELEM * C_CH * MIJ) return;
    int mij = t % MIJ;
    int c   = (t / MIJ) % C_CH;
    int e   = t / (MIJ * C_CH);
    float v = 0.f;
    if (mij < 81) {
        int ij = mij;
        for (int p = 0; p < 2; ++p)
            v += U2_0[ij * 2 + p] * W2_0[(e * 2 + p) * C_CH + c];
    } else {
        int q = mij - 81;
        int m = q / 81, ij = q % 81;
        for (int p = 0; p < 2; ++p)
            v += U2_1[(ij * 3 + m) * 2 + p] * W2_1[(e * 2 + p) * C_CH + c];
    }
    c2w[t] = v;
}

// K1c: C1w[e,c,mi] = U1[i,m,0] * W1[e,0,c]
__global__ __launch_bounds__(256)
void k1_c1(const float* __restrict__ U1_0, const float* __restrict__ W1_0,
           const float* __restrict__ U1_1, const float* __restrict__ W1_1,
           float* __restrict__ c1w) {
    int t = blockIdx.x * blockDim.x + threadIdx.x;
    if (t >= E_ELEM * C_CH * MI_TOT) return;
    int mi = t % MI_TOT;
    int c  = (t / MI_TOT) % C_CH;
    int e  = t / (MI_TOT * C_CH);
    float v;
    if (mi < 9) {
        v = U1_0[mi] * W1_0[e * C_CH + c];
    } else {
        int m = (mi - 9) / 9, i = (mi - 9) % 9;
        v = U1_1[i * 3 + m] * W1_1[e * C_CH + c];
    }
    c1w[t] = v;
}

// ---------------------------------------------------------------------------
// K2: main contraction. Grid (74 node-tiles, 128 channels), 256 thr = 8 waves.
//   Stage 1: D[336 mij x 16 nodes] via 21 WMMA_F32_16X16X32_BF16
//            (unrolled: 2 unconditional + 1 wave-uniform-guarded per wave).
//   Stage 2: per-(node,i) partials by 144 threads into LDS.
//   Stage 3: 9-way reduction + y write by 64 threads.
// x rows staged with async global->LDS copies (ASYNCcnt path).
// ---------------------------------------------------------------------------
__global__ __launch_bounds__(256)
void k2_main(const float* __restrict__ nf, const unsigned short* __restrict__ apack,
             const float* __restrict__ c2w, const float* __restrict__ c1w,
             const int* __restrict__ meta, float* __restrict__ yt) {
    __shared__ float xsh[16][K_SPH];
    __shared__ int   idx[16];
    __shared__ __align__(32) unsigned short Bsh[32 * 16];
    __shared__ float Dsh[MIJ_T * 16][16];            // 336 x 16 f32 (~21.5 KB)
    __shared__ float Psh[4][K_SPH][16];              // per-(m,i,node) partials

    int T = blockIdx.x;
    if (T >= meta[0]) return;                        // uniform: no divergent barriers
    int c    = blockIdx.y;
    int e    = meta[16 + T];
    int base = meta[96 + T];
    const int* list = meta + 176;
    int tid = threadIdx.x;

    if (tid < 16) idx[tid] = list[base + tid];
    __syncthreads();

    // --- gather x rows: async global->LDS for valid nodes, DS zeros for pads.
    if (tid < 16 * K_SPH) {
        int nn = tid / K_SPH, k = tid % K_SPH;
        int nd = idx[nn];
        if (nd >= 0) {
            const float* g = nf + ((size_t)nd * C_CH + c) * K_SPH + k;
            unsigned lofs = (unsigned)(size_t)&xsh[nn][k];   // LDS dest address
            asm volatile("global_load_async_to_lds_b32 %0, %1, off"
                         :: "v"(lofs), "v"(g) : "memory");
        } else {
            xsh[nn][k] = 0.f;                        // disjoint from async writes
        }
    }
    asm volatile("s_wait_asynccnt 0x0" ::: "memory");
    __syncthreads();

    if (tid < 32) {                                  // B layout: lane%16=col, K=table order
        int lane = tid;
        for (int s = 0; s < 16; ++s) {
            int k = (lane < 16) ? ((s < 8) ? s : 16 + (s - 8)) : ((s < 8) ? 8 + s : 24 + (s - 8));
            float v = (k < K_SPH) ? xsh[lane & 15][k] : 0.f;
            Bsh[lane * 16 + s] = f2bf(v);
        }
    }
    __syncthreads();

    int lane = tid & 31;
    int wave = tid >> 5;
    const v16bf b = *reinterpret_cast<const v16bf*>(&Bsh[lane * 16]);
    const unsigned short* abase = apack + ((size_t)(e * C_CH + c) * MIJ_T) * 32 * 16;

    // Tiles wave, wave+8 always exist (wave+8 <= 15 < 21); wave+16 for waves 0-4.
    const bool has2 = (wave + 16) < MIJ_T;           // wave-uniform
    const unsigned short* ap0 = abase + ((size_t)(wave     ) * 32 + lane) * 16;
    const unsigned short* ap1 = abase + ((size_t)(wave +  8) * 32 + lane) * 16;
    const unsigned short* ap2 = abase + ((size_t)(wave + 16) * 32 + lane) * 16;
    v16bf a0 = *reinterpret_cast<const v16bf*>(ap0);
    v16bf a1 = *reinterpret_cast<const v16bf*>(ap1);
    v16bf a2 = {};
    if (has2) a2 = *reinterpret_cast<const v16bf*>(ap2);

    int col   = lane & 15;
    int rbase = (lane >> 4) * 8;                     // C/D: lanes>=16 -> rows 8..15
    v8f z = {};
    v8f d0 = __builtin_amdgcn_wmma_f32_16x16x32_bf16(false, a0, false, b, (short)0, z, false, false);
    #pragma unroll
    for (int r = 0; r < 8; ++r) Dsh[wave * 16 + rbase + r][col] = d0[r];
    v8f d1 = __builtin_amdgcn_wmma_f32_16x16x32_bf16(false, a1, false, b, (short)0, z, false, false);
    #pragma unroll
    for (int r = 0; r < 8; ++r) Dsh[(wave + 8) * 16 + rbase + r][col] = d1[r];
    if (has2) {
        v8f d2 = __builtin_amdgcn_wmma_f32_16x16x32_bf16(false, a2, false, b, (short)0, z, false, false);
        #pragma unroll
        for (int r = 0; r < 8; ++r) Dsh[(wave + 16) * 16 + rbase + r][col] = d2[r];
    }
    __syncthreads();

    // --- stage 2: partials over (node nn, component i) for all 4 m slices.
    if (tid < 16 * K_SPH) {
        int nn = tid & 15, i = tid >> 4;             // i in 0..8
        const float* c2 = c2w + (size_t)(e * C_CH + c) * MIJ;
        const float* c1 = c1w + (size_t)(e * C_CH + c) * MI_TOT;
        float xi = xsh[nn][i];
        #pragma unroll
        for (int m = 0; m < 4; ++m) {
            int mijBase = (m == 0) ? 0 : 81 + (m - 1) * 81;
            int miBase  = (m == 0) ? 0 : 9 + (m - 1) * 9;
            int rb = mijBase + i * 9;
            float ct = c1[miBase + i];
            #pragma unroll
            for (int j = 0; j < K_SPH; ++j)
                ct += (Dsh[rb + j][nn] + c2[rb + j]) * xsh[nn][j];
            Psh[m][i][nn] = ct * xi;
        }
    }
    __syncthreads();

    // --- stage 3: reduce over i and write y[n,c,m].
    if (tid < 64) {
        int nn = tid & 15, m = tid >> 4;
        int nd = idx[nn];
        if (nd >= 0) {
            float y = 0.f;
            #pragma unroll
            for (int i = 0; i < K_SPH; ++i) y += Psh[m][i][nn];
            yt[((size_t)nd * C_CH + c) * 4 + m] = y;
        }
    }
}

// ---------------------------------------------------------------------------
// K3: e3nn Linear as bf16 WMMA GEMM: z[n,d,m] = inv * sum_c y[n,c,m]*Wlin[c,d]
// Grid (64 node-tiles, 4 m-slices); 8 waves each own one 16-wide d tile.
// Fully unrolled K = 4 x 32 -> 4 static WMMAs chained through the C operand.
// ---------------------------------------------------------------------------
__global__ __launch_bounds__(256)
void k3_linear(const float* __restrict__ yt, const float* __restrict__ wlin0,
               const float* __restrict__ wlin1, const float* __restrict__ sc,
               float* __restrict__ out) {
    __shared__ __align__(32) unsigned short Ysh[16 * 32];
    int n0 = blockIdx.x * 16;
    int m  = blockIdx.y;                              // 0 -> L0, 1..3 -> L1
    const float* wl = (m == 0) ? wlin0 : wlin1;
    int tid  = threadIdx.x;
    int lane = tid & 31;
    int dt   = tid >> 5;                              // wave id = d tile

    v8f acc = {};
    #pragma unroll
    for (int kt = 0; kt < 4; ++kt) {                  // K = 128 = 4 x 32
        __syncthreads();
        for (int i = tid; i < 16 * 32; i += 256) {    // stage y tile -> bf16
            int r = i >> 5, cc = i & 31;
            float v = yt[((size_t)(n0 + r) * C_CH + (kt * 32 + cc)) * 4 + m];
            Ysh[r * 32 + cc] = f2bf(v);
        }
        __syncthreads();
        union { unsigned short u[16]; v16bf v; } A, B;
        int arow = lane & 15;
        #pragma unroll
        for (int s = 0; s < 16; ++s) {
            int k = (lane < 16) ? ((s < 8) ? s : 16 + (s - 8))
                                : ((s < 8) ? 8 + s : 24 + (s - 8));
            A.u[s] = Ysh[arow * 32 + k];
        }
        int dg = dt * 16 + (lane & 15);
        #pragma unroll
        for (int s = 0; s < 16; ++s) {
            int k = (lane < 16) ? ((s < 8) ? s : 16 + (s - 8))
                                : ((s < 8) ? 8 + s : 24 + (s - 8));
            B.u[s] = f2bf(wl[(size_t)(kt * 32 + k) * C_CH + dg]);
        }
        acc = __builtin_amdgcn_wmma_f32_16x16x32_bf16(false, A.v, false, B.v,
                                                      (short)0, acc, false, false);
    }

    const float inv = 0.08838834764831845f;           // 1/sqrt(128)
    int dg    = dt * 16 + (lane & 15);
    int rbase = (lane >> 4) * 8;
    #pragma unroll
    for (int r = 0; r < 8; ++r) {
        int n  = n0 + rbase + r;
        int oc = (m == 0) ? dg : (128 + dg * 3 + (m - 1));
        out[(size_t)n * 512 + oc] = acc[r] * inv + sc[(size_t)n * 512 + oc];
    }
}

// ---------------------------------------------------------------------------
extern "C" void kernel_launch(void* const* d_in, const int* in_sizes, int n_in,
                              void* d_out, int out_size, void* d_ws, size_t ws_size,
                              hipStream_t stream) {
    (void)in_sizes; (void)n_in; (void)out_size; (void)ws_size;
    const float* nf    = (const float*)d_in[0];
    const float* na    = (const float*)d_in[1];
    const float* sc    = (const float*)d_in[2];
    const float* U1_0  = (const float*)d_in[3];
    const float* W1_0  = (const float*)d_in[4];
    const float* U2_0  = (const float*)d_in[5];
    const float* W2_0  = (const float*)d_in[6];
    const float* U3_0  = (const float*)d_in[7];
    const float* W3_0  = (const float*)d_in[8];
    const float* Wlin0 = (const float*)d_in[9];
    const float* U1_1  = (const float*)d_in[10];
    const float* W1_1  = (const float*)d_in[11];
    const float* U2_1  = (const float*)d_in[12];
    const float* W2_1  = (const float*)d_in[13];
    const float* U3_1  = (const float*)d_in[14];
    const float* W3_1  = (const float*)d_in[15];
    const float* Wlin1 = (const float*)d_in[16];

    char* ws = (char*)d_ws;
    unsigned short* apack = (unsigned short*)(ws + APACK_OFF);
    float* c2w = (float*)(ws + C2W_OFF);
    float* c1w = (float*)(ws + C1W_OFF);
    float* yt  = (float*)(ws + Y_OFF);
    int*   meta = (int*)(ws + META_OFF);
    float* out = (float*)d_out;

    k0_group<<<dim3(1), dim3(1024), 0, stream>>>(na, meta);
    k1_packA<<<dim3((E_ELEM * C_CH * MIJ_T * 32) / 256), dim3(256), 0, stream>>>(
        U3_0, W3_0, U3_1, W3_1, apack);
    k1_c2<<<dim3((E_ELEM * C_CH * MIJ) / 256), dim3(256), 0, stream>>>(
        U2_0, W2_0, U2_1, W2_1, c2w);
    k1_c1<<<dim3((E_ELEM * C_CH * MI_TOT) / 256), dim3(256), 0, stream>>>(
        U1_0, W1_0, U1_1, W1_1, c1w);
    k2_main<<<dim3(74, C_CH), dim3(256), 0, stream>>>(nf, apack, c2w, c1w, meta, yt);
    k3_linear<<<dim3(N_NODES / 16, 4), dim3(256), 0, stream>>>(yt, Wlin0, Wlin1, sc, out);
}
